// PI_STGNN_49314814493242
// MI455X (gfx1250) — matrix operand order repaired and checked by reading, not compiled, not run on hardware
//
#include <hip/hip_runtime.h>
#include <hip/hip_bf16.h>
#include <math.h>

// ---------- problem constants ----------
#define Bq 8
#define Tq 32
#define Nn 10000
#define Ff 64
#define Hh 128
#define Ee 160000
#define NTILES 625          // Nn / 16 (exact)
#define ROWB 144            // LDS row stride in bytes for 16x64 bf16 tile (64*2 + pad, 16B aligned)

typedef __attribute__((ext_vector_type(16))) __bf16 v16bf;
typedef __attribute__((ext_vector_type(8)))  float  v8f;

union AFrag { uint4 q[2]; v16bf v; };

__device__ __forceinline__ float sigf(float x) { return 1.0f / (1.0f + __expf(-x)); }

// ---------- 1) init workspace: deg=1 (self loop), counts=0, cursor=0, emb=0 ----------
__global__ void k_init(float* __restrict__ degf, int* __restrict__ counts,
                       int* __restrict__ cursor, float* __restrict__ emb) {
    int i = blockIdx.x * blockDim.x + threadIdx.x;
    if (i < Nn) { degf[i] = 1.0f; counts[i] = 0; cursor[i] = 0; }
    if (i < Bq * Tq * Hh) emb[i] = 0.0f;
}

// ---------- 2) edge counting (in-degree on dst) ----------
__global__ void k_count(const int* __restrict__ ei, float* __restrict__ degf,
                        int* __restrict__ counts) {
    int e = blockIdx.x * blockDim.x + threadIdx.x;
    if (e >= Ee) return;
    int d = ei[Ee + e];                 // dst row of edge_index [2,E]
    atomicAdd(&degf[d], 1.0f);
    atomicAdd(&counts[d], 1);
}

// ---------- 3) dinv = deg^-1/2 ----------
__global__ void k_dinv(const float* __restrict__ degf, float* __restrict__ dinv) {
    int i = blockIdx.x * blockDim.x + threadIdx.x;
    if (i < Nn) dinv[i] = rsqrtf(degf[i]);   // deg >= 1 always (self loop)
}

// ---------- 4) exclusive scan of counts -> rowptr (single workgroup) ----------
__global__ __launch_bounds__(1024) void k_scan(const int* __restrict__ counts,
                                               int* __restrict__ rowptr) {
    __shared__ int part[1024];
    const int CH = 10;                  // 1024*10 >= Nn
    int tid = threadIdx.x;
    int base = tid * CH;
    int s = 0;
    for (int j = 0; j < CH; ++j) { int idx = base + j; if (idx < Nn) s += counts[idx]; }
    part[tid] = s;
    __syncthreads();
    for (int off = 1; off < 1024; off <<= 1) {
        int v = (tid >= off) ? part[tid - off] : 0;
        __syncthreads();
        part[tid] += v;
        __syncthreads();
    }
    int run = part[tid] - s;            // exclusive prefix
    for (int j = 0; j < CH; ++j) {
        int idx = base + j;
        if (idx < Nn) { rowptr[idx] = run; run += counts[idx]; }
    }
    if (tid == 1023) rowptr[Nn] = part[1023];
}

// ---------- 5) CSR fill: colidx = src, vals = dinv[src]*dinv[dst] ----------
__global__ void k_fill(const int* __restrict__ ei, const int* __restrict__ rowptr,
                       int* __restrict__ cursor, const float* __restrict__ dinv,
                       int* __restrict__ colidx, float* __restrict__ vals) {
    int e = blockIdx.x * blockDim.x + threadIdx.x;
    if (e >= Ee) return;
    int s = ei[e];
    int d = ei[Ee + e];
    int pos = rowptr[d] + atomicAdd(&cursor[d], 1);
    colidx[pos] = s;
    vals[pos]   = dinv[s] * dinv[d];
}

// ---------- 6) pack W[64,128] f32 -> bf16 in WMMA B-fragment order ----------
// wpack[(((ct*2+kt)*32 + lane)*16 + j)] = bf16(W[kt*32 + (lane>>4)*16 + j][ct*16 + (lane&15)])
__global__ void k_packw(const float* __restrict__ w_gcn, __bf16* __restrict__ wpack) {
    int idx = blockIdx.x * blockDim.x + threadIdx.x;
    if (idx >= 8 * 2 * 32 * 16) return;
    int j    = idx & 15;
    int lane = (idx >> 4) & 31;
    int kt   = (idx >> 9) & 1;
    int ct   = idx >> 10;
    int k = kt * 32 + (lane >> 4) * 16 + j;
    int n = ct * 16 + (lane & 15);
    wpack[idx] = (__bf16)w_gcn[k * Hh + n];
}

// ---------- 7) fused GCN: CSR-aggregate 16 nodes -> LDS bf16 -> WMMA GEMM -> ReLU/mean ----------
__global__ __launch_bounds__(128) void k_gcn(const float* __restrict__ x_seq,
                                             const int* __restrict__ rowptr,
                                             const int* __restrict__ colidx,
                                             const float* __restrict__ vals,
                                             const float* __restrict__ dinv,
                                             const __bf16* __restrict__ wpack,
                                             const float* __restrict__ bgcn,
                                             float* __restrict__ emb) {
    const int bt = blockIdx.y;
    const int m0 = blockIdx.x * 16;
    const float* __restrict__ x = x_seq + (size_t)bt * Nn * Ff;
    __shared__ __align__(16) unsigned char tileA[16 * ROWB];

    const int tid = threadIdx.x;

    // ---- aggregation: 8 threads per node, 8 features each (feature-side, F=64) ----
    {
        int sub = tid >> 3;            // node within tile 0..15
        int fl  = tid & 7;             // feature group 0..7 (8 floats each)
        int node = m0 + sub;
        float dd = dinv[node];
        float selfw = dd * dd;
        const float* xr = x + (size_t)node * Ff + fl * 8;
        float4 a0 = *(const float4*)(xr);
        float4 a1 = *(const float4*)(xr + 4);
        float acc[8] = { selfw * a0.x, selfw * a0.y, selfw * a0.z, selfw * a0.w,
                         selfw * a1.x, selfw * a1.y, selfw * a1.z, selfw * a1.w };
        int pbeg = rowptr[node], pend = rowptr[node + 1];
        for (int p = pbeg; p < pend; ++p) {
            int s  = colidx[p];
            float v = vals[p];
            const float* xs = x + (size_t)s * Ff + fl * 8;
            float4 b0 = *(const float4*)(xs);
            float4 b1 = *(const float4*)(xs + 4);
            acc[0] += v * b0.x; acc[1] += v * b0.y; acc[2] += v * b0.z; acc[3] += v * b0.w;
            acc[4] += v * b1.x; acc[5] += v * b1.y; acc[6] += v * b1.z; acc[7] += v * b1.w;
        }
        __bf16* dstp = (__bf16*)(tileA + sub * ROWB + fl * 16);
#pragma unroll
        for (int j = 0; j < 8; ++j) dstp[j] = (__bf16)acc[j];
    }
    __syncthreads();

    // ---- WMMA: 4 waves x 2 column tiles; K=64 -> 2 x K32 bf16 WMMAs ----
    const int lane = tid & 31;
    const int wave = tid >> 5;
    const int row  = lane & 15;
    const int kh   = lane >> 4;

    AFrag af[2];
#pragma unroll
    for (int kt = 0; kt < 2; ++kt) {
        int kb = kt * 32 + kh * 16;    // A element j -> K=kb+j (contiguous 16)
        af[kt].q[0] = *(const uint4*)(tileA + row * ROWB + kb * 2);
        af[kt].q[1] = *(const uint4*)(tileA + row * ROWB + kb * 2 + 16);
    }

#pragma unroll
    for (int i = 0; i < 2; ++i) {
        int ct = wave * 2 + i;
        int n0 = ct * 16;
        v8f c = {};
#pragma unroll
        for (int kt = 0; kt < 2; ++kt) {
            AFrag bf_;
            const uint4* wp = (const uint4*)wpack + ((size_t)((ct * 2 + kt) * 32 + lane)) * 2;
            bf_.q[0] = wp[0];
            bf_.q[1] = wp[1];
            c = __builtin_amdgcn_wmma_f32_16x16x32_bf16(
                    false, af[kt].v, false, bf_.v, (short)0, c, false, false);
        }
        // epilogue: bias + relu + column sum over 16 rows -> mean accumulation
        int col = lane & 15;
        float bias = bgcn[n0 + col];
        float s = 0.0f;
#pragma unroll
        for (int v = 0; v < 8; ++v) {
            float t = c[v] + bias;
            s += (t > 0.0f) ? t : 0.0f;
        }
        s += __shfl_xor(s, 16, 32);    // combine rows 0-7 and 8-15 halves (wave32)
        if (lane < 16)
            atomicAdd(&emb[(size_t)bt * Hh + n0 + col], s * (1.0f / (float)Nn));
    }
}

// ---------- 8) LSTM (32 steps, batch 8, H=128) + classifier head ----------
__global__ __launch_bounds__(512) void k_lstm(const float* __restrict__ emb,
                                              const float* __restrict__ w_ih,
                                              const float* __restrict__ w_hh,
                                              const float* __restrict__ b_ih,
                                              const float* __restrict__ b_hh,
                                              const float* __restrict__ w_cls,
                                              const float* __restrict__ b_cls,
                                              float* __restrict__ out) {
    __shared__ float hb[Bq][Hh];
    __shared__ float cb[Bq][Hh];
    __shared__ float xs[Bq][Hh];
    __shared__ float gates[Bq][4 * Hh];
    int tid = threadIdx.x;

    for (int i = tid; i < Bq * Hh; i += 512) { hb[i >> 7][i & 127] = 0.0f; cb[i >> 7][i & 127] = 0.0f; }
    __syncthreads();

    for (int t = 0; t < Tq; ++t) {
        for (int i = tid; i < Bq * Hh; i += 512) {
            int b = i >> 7, j = i & 127;
            xs[b][j] = emb[((size_t)b * Tq + t) * Hh + j];
        }
        __syncthreads();

        // each thread computes one of the 512 gate rows for all 8 batches
        {
            int g = tid;
            float bb = b_ih[g] + b_hh[g];
            float d[Bq];
#pragma unroll
            for (int b = 0; b < Bq; ++b) d[b] = bb;
            for (int k = 0; k < Hh; ++k) {
                float wi = w_ih[g * Hh + k];
                float wh = w_hh[g * Hh + k];
#pragma unroll
                for (int b = 0; b < Bq; ++b) d[b] += wi * xs[b][k] + wh * hb[b][k];
            }
#pragma unroll
            for (int b = 0; b < Bq; ++b) gates[b][g] = d[b];
        }
        __syncthreads();

        for (int i = tid; i < Bq * Hh; i += 512) {
            int b = i >> 7, j = i & 127;
            float ig = gates[b][j];
            float fg = gates[b][Hh + j];
            float gg = gates[b][2 * Hh + j];
            float og = gates[b][3 * Hh + j];
            float cc = sigf(fg) * cb[b][j] + sigf(ig) * tanhf(gg);
            cb[b][j] = cc;
            hb[b][j] = sigf(og) * tanhf(cc);
        }
        __syncthreads();
    }

    if (tid < Bq) {
        float s = b_cls[0];
        for (int j = 0; j < Hh; ++j) s += hb[tid][j] * w_cls[j];
        out[tid] = 1.0f / (1.0f + __expf(-s));
    }
}

// ---------- launcher ----------
extern "C" void kernel_launch(void* const* d_in, const int* in_sizes, int n_in,
                              void* d_out, int out_size, void* d_ws, size_t ws_size,
                              hipStream_t stream) {
    const float* x_seq = (const float*)d_in[0];
    const int*   ei    = (const int*)  d_in[1];
    const float* w_gcn = (const float*)d_in[2];
    const float* b_gcn = (const float*)d_in[3];
    const float* w_ih  = (const float*)d_in[4];
    const float* w_hh  = (const float*)d_in[5];
    const float* b_ih  = (const float*)d_in[6];
    const float* b_hh  = (const float*)d_in[7];
    const float* w_cls = (const float*)d_in[8];
    const float* b_cls = (const float*)d_in[9];
    float* out = (float*)d_out;

    // workspace carve-out (256B aligned), ~1.63 MB total
    char* base = (char*)d_ws;
    size_t off = 0;
    auto carve = [&](size_t bytes) -> char* {
        char* p = base + off;
        off = (off + bytes + 255) & ~(size_t)255;
        return p;
    };
    float* degf   = (float*)carve(Nn * sizeof(float));
    float* dinv   = (float*)carve(Nn * sizeof(float));
    int*   counts = (int*)  carve(Nn * sizeof(int));
    int*   rowptr = (int*)  carve((Nn + 1) * sizeof(int));
    int*   cursor = (int*)  carve(Nn * sizeof(int));
    int*   colidx = (int*)  carve(Ee * sizeof(int));
    float* vals   = (float*)carve(Ee * sizeof(float));
    __bf16* wpack = (__bf16*)carve(8 * 2 * 32 * 16 * sizeof(__bf16));
    float* emb    = (float*)carve(Bq * Tq * Hh * sizeof(float));
    (void)ws_size; (void)n_in; (void)in_sizes; (void)out_size;

    const int TB = 256;
    int initN = (Bq * Tq * Hh > Nn) ? Bq * Tq * Hh : Nn;

    k_init <<<(initN + TB - 1) / TB, TB, 0, stream>>>(degf, counts, cursor, emb);
    k_count<<<(Ee + TB - 1) / TB, TB, 0, stream>>>(ei, degf, counts);
    k_dinv <<<(Nn + TB - 1) / TB, TB, 0, stream>>>(degf, dinv);
    k_scan <<<1, 1024, 0, stream>>>(counts, rowptr);
    k_fill <<<(Ee + TB - 1) / TB, TB, 0, stream>>>(ei, rowptr, cursor, dinv, colidx, vals);
    k_packw<<<(8192 + TB - 1) / TB, TB, 0, stream>>>(w_gcn, wpack);

    dim3 grid(NTILES, Bq * Tq);   // 625 x 256 blocks, 128 threads (4 waves) each
    k_gcn<<<grid, 128, 0, stream>>>(x_seq, rowptr, colidx, vals, dinv, wpack, b_gcn, emb);

    k_lstm<<<1, 512, 0, stream>>>(emb, w_ih, w_hh, b_ih, b_hh, w_cls, b_cls, out);
}
